// Conditional_forest_49340584296938
// MI455X (gfx1250) — matrix operand order repaired and checked by reading, not compiled, not run on hardware
//
#include <hip/hip_runtime.h>

typedef __attribute__((ext_vector_type(16))) _Float16 v16h;
typedef __attribute__((ext_vector_type(8)))  _Float16 v8h;
typedef __attribute__((ext_vector_type(4)))  _Float16 v4h;
typedef __attribute__((ext_vector_type(8)))  float    v8f;

#define B_DIM   32
#define T_DIM   2048
#define D_DIM   512
#define K_TREES 8
#define L_DIM   16
#define N_NODE  255
#define N_LEAF  256
#define N_PAD   256
#define ROWS    (B_DIM * T_DIM)   // 65536
#define ROW_TILES (ROWS / 16)     // 4096
#define XS_STRIDE 520             // 512 + 8 halfs: bank-spread padding

// ---------- prep 1: W (K,255,512) f32 -> Wh (K,256,512) f16, node 255 zero-padded
__global__ __launch_bounds__(256) void prep_w(const float* __restrict__ W,
                                              _Float16* __restrict__ Wh) {
    int idx  = blockIdx.x * 256 + threadIdx.x;   // over K*256*512 = 2^20
    int f    = idx & 511;
    int node = (idx >> 9) & 255;
    int k    = idx >> 17;
    float v  = (node < N_NODE) ? W[((size_t)k * N_NODE + node) * D_DIM + f] : 0.0f;
    Wh[idx]  = (_Float16)v;
}

// ---------- prep 2: P = softmax(pi, axis=-1), rows of L=16
__global__ __launch_bounds__(256) void prep_p(const float* __restrict__ pi,
                                              float* __restrict__ P) {
    int row = blockIdx.x * 256 + threadIdx.x;    // 0 .. K*256-1 = 2047
    const float4* src = (const float4*)(pi + (size_t)row * L_DIM);
    float v[16];
#pragma unroll
    for (int q = 0; q < 4; ++q) {
        float4 t = src[q];
        v[q*4+0] = t.x; v[q*4+1] = t.y; v[q*4+2] = t.z; v[q*4+3] = t.w;
    }
    float m = -1e30f;
#pragma unroll
    for (int l = 0; l < 16; ++l) m = fmaxf(m, v[l]);
    float s = 0.0f;
#pragma unroll
    for (int l = 0; l < 16; ++l) { v[l] = __expf(v[l] - m); s += v[l]; }
    float inv = 1.0f / s;
    float4* dst = (float4*)(P + (size_t)row * L_DIM);
#pragma unroll
    for (int q = 0; q < 4; ++q) {
        float4 t;
        t.x = v[q*4+0]*inv; t.y = v[q*4+1]*inv; t.z = v[q*4+2]*inv; t.w = v[q*4+3]*inv;
        dst[q] = t;
    }
}

// ---------- main fused kernel: WMMA GEMM -> sigmoid -> tree mu -> tp = mu * P
// block = 256 threads (8 waves), handles 16 rows of (b,t); loops k = 0..7.
// tp laid out t-major: tp[((t*B + b)*K + k)*L + l]  for the serial scan.
__global__ __launch_bounds__(256) void forest_tp(
        const float* __restrict__ x, const _Float16* __restrict__ Wh,
        const float* __restrict__ bias, const float* __restrict__ P,
        float* __restrict__ tp)
{
    __shared__ _Float16 xs[16 * XS_STRIDE];   // 16.25 KB, f16 x tile (padded stride)
    __shared__ float    ds[16 * 256];         // 16 KB, sigmoid(d) for one tree
    __shared__ float    Ps[256 * 16];         // 16 KB, softmax(pi[k])
    __shared__ _Float16 mus[16 * 256];        // 8 KB, leaf probabilities

    const int rt   = blockIdx.x;      // row tile 0..4095
    const int tid  = threadIdx.x;
    const int wave = tid >> 5;
    const int lane = tid & 31;
    const int hi   = lane >> 4;       // K-half selector (A/B frag layout)
    const int lo   = lane & 15;       // row (A) / col (B) within tile

    // ---- stage x tile (16 rows x 512 f32) -> f16 LDS, padded stride
    {
        const float4* xin = (const float4*)(x + (size_t)rt * 16 * D_DIM);
#pragma unroll
        for (int i = 0; i < 8; ++i) {
            int idx  = tid + i * 256;     // float4 index, 0..2047
            int row  = idx >> 7;          // 128 float4 per row
            int col4 = idx & 127;
            float4 v = xin[idx];
            v4h h;
            h[0] = (_Float16)v.x; h[1] = (_Float16)v.y;
            h[2] = (_Float16)v.z; h[3] = (_Float16)v.w;
            *(v4h*)(xs + row * XS_STRIDE + col4 * 4) = h;
        }
    }
    __syncthreads();

    for (int k = 0; k < K_TREES; ++k) {
        // stage softmax(pi) for this tree (consumed after the next barrier)
        {
            const float4* pin = (const float4*)(P + (size_t)k * N_LEAF * L_DIM);
            float4* pls = (float4*)Ps;
#pragma unroll
            for (int i = 0; i < 4; ++i) pls[tid + i * 256] = pin[tid + i * 256];
        }
        // ---- GEMM: wave handles node tiles nt = 2*wave, 2*wave+1
#pragma unroll
        for (int nti = 0; nti < 2; ++nti) {
            const int nt = wave * 2 + nti;
            v8f acc = {};
            const _Float16* wrow = Wh + ((size_t)k * N_PAD + nt * 16 + lo) * D_DIM + hi * 16;
            const _Float16* arow = xs + lo * XS_STRIDE + hi * 16;
#pragma unroll
            for (int kk = 0; kk < 16; ++kk) {
                union { v16h v; v8h h[2]; } a;
                a.h[0] = *(const v8h*)(arow + kk * 32);        // K = 16*hi + 0..7
                a.h[1] = *(const v8h*)(arow + kk * 32 + 8);    // K = 16*hi + 8..15
                v16h bf = *(const v16h*)(wrow + kk * 32);      // K = 16*hi + 0..15, col lo
                acc = __builtin_amdgcn_wmma_f32_16x16x32_f16(
                        false, a.v, false, bf, (short)0, acc, false, false);
            }
            const int node = nt * 16 + lo;
            const float bv = (node < N_NODE) ? bias[k * N_NODE + node] : 0.0f;
#pragma unroll
            for (int r = 0; r < 8; ++r) {
                const int row = r + hi * 8;
                float z = acc[r] + bv;
                ds[row * 256 + node] = 1.0f / (1.0f + __expf(-z));
            }
        }
        __syncthreads();
        // ---- mu phase: thread = (row, leaf-group of 16)
        {
            const int row = tid >> 4;
            const int lg  = tid & 15;
#pragma unroll
            for (int jj = 0; jj < 16; ++jj) {
                const int j = lg * 16 + jj;
                float mu = 1.0f;
#pragma unroll
                for (int l = 0; l < 8; ++l) {
                    const int node = (1 << l) - 1 + (j >> (8 - l));
                    const float dv = ds[row * 256 + node];
                    mu *= ((j >> (7 - l)) & 1) ? (1.0f - dv) : dv;
                }
                mus[row * 256 + j] = (_Float16)mu;
            }
        }
        __syncthreads();
        // ---- tp phase: thread = (row, l): tp = sum_j mu[row,j] * P[j,l]
        {
            const int row = tid >> 4;
            const int l   = tid & 15;
            float acc2 = 0.0f;
#pragma unroll 8
            for (int j = 0; j < 256; ++j)
                acc2 = fmaf((float)mus[row * 256 + j], Ps[j * 16 + l], acc2);
            const int gr = rt * 16 + row;          // global (b,t) row
            const int bb = gr >> 11;               // / T_DIM
            const int t  = gr & (T_DIM - 1);
            tp[(((size_t)t * B_DIM + bb) * K_TREES + k) * L_DIM + l] = acc2;
        }
        __syncthreads();   // ds / Ps reused next k
    }
}

// ---------- serial scan: single wave32, lane b owns batch row b entirely.
// tmp_{t+1} = softmax(exp(tp_t^T tmp_t)); preds[t] = tp_t^T tmp_t  (pre-activation)
__global__ __launch_bounds__(32) void forest_scan(const float* __restrict__ tp,
                                                  float* __restrict__ out) {
    const int b = threadIdx.x;
    float tmp[16];
#pragma unroll
    for (int l = 0; l < 16; ++l) tmp[l] = 1.0f / 16.0f;
    float* outb = out + (size_t)b * T_DIM * L_DIM;

    for (int t = 0; t < T_DIM; ++t) {
        const float4* base = (const float4*)(tp + ((size_t)t * B_DIM + b) * (K_TREES * L_DIM));
        // hide HBM latency: prefetch this lane's 512B slice 6 iterations ahead
        if (t + 6 < T_DIM) {
            const char* pf = (const char*)(tp + ((size_t)(t + 6) * B_DIM + b) * (K_TREES * L_DIM));
            __builtin_prefetch(pf +   0, 0, 0);
            __builtin_prefetch(pf + 128, 0, 0);
            __builtin_prefetch(pf + 256, 0, 0);
            __builtin_prefetch(pf + 384, 0, 0);
        }
        float nv[16];
#pragma unroll
        for (int l = 0; l < 16; ++l) nv[l] = 0.0f;
#pragma unroll
        for (int k = 0; k < 8; ++k) {
            const float tk = tmp[k];
#pragma unroll
            for (int q = 0; q < 4; ++q) {
                float4 v = base[k * 4 + q];
                nv[q*4+0] = fmaf(v.x, tk, nv[q*4+0]);
                nv[q*4+1] = fmaf(v.y, tk, nv[q*4+1]);
                nv[q*4+2] = fmaf(v.z, tk, nv[q*4+2]);
                nv[q*4+3] = fmaf(v.w, tk, nv[q*4+3]);
            }
        }
        float4* o4 = (float4*)(outb + (size_t)t * L_DIM);
#pragma unroll
        for (int q = 0; q < 4; ++q) {
            float4 v;
            v.x = nv[q*4+0]; v.y = nv[q*4+1]; v.z = nv[q*4+2]; v.w = nv[q*4+3];
            o4[q] = v;
        }
        // tmp = softmax(exp(nv))
        float e1[16];
        float m = -1e30f;
#pragma unroll
        for (int l = 0; l < 16; ++l) { e1[l] = __expf(nv[l]); m = fmaxf(m, e1[l]); }
        float s = 0.0f;
#pragma unroll
        for (int l = 0; l < 16; ++l) { float e = __expf(e1[l] - m); tmp[l] = e; s += e; }
        float inv = 1.0f / s;
#pragma unroll
        for (int l = 0; l < 16; ++l) tmp[l] *= inv;
    }
}

extern "C" void kernel_launch(void* const* d_in, const int* in_sizes, int n_in,
                              void* d_out, int out_size, void* d_ws, size_t ws_size,
                              hipStream_t stream) {
    const float* x  = (const float*)d_in[0];
    const float* W  = (const float*)d_in[1];
    const float* bs = (const float*)d_in[2];
    const float* pi = (const float*)d_in[3];
    // inputs 4,5 are conditional=1, dependence_length=-1 (fixed by setup)

    char* ws = (char*)d_ws;
    _Float16* Wh = (_Float16*)ws;                              // 2 MB
    float*    P  = (float*)(ws + (2u << 20));                  // 128 KB
    float*    tp = (float*)(ws + (2u << 20) + (1u << 18));     // 32 MB (t-major)

    prep_w<<<(K_TREES * N_PAD * D_DIM) / 256, 256, 0, stream>>>(W, Wh);
    prep_p<<<(K_TREES * N_LEAF) / 256, 256, 0, stream>>>(pi, P);
    forest_tp<<<ROW_TILES, 256, 0, stream>>>(x, Wh, bs, P, tp);
    forest_scan<<<1, 32, 0, stream>>>(tp, (float*)d_out);
}
